// GraniteMoeMoE_49435073577023
// MI455X (gfx1250) — compile-verified, matrix operand order checked by reading, not run on hardware
//
#include <hip/hip_runtime.h>
#include <math.h>

typedef __bf16 bf16;
typedef __attribute__((ext_vector_type(16))) __bf16 v16bf;
typedef __attribute__((ext_vector_type(8)))  __bf16 v8bf;
typedef __attribute__((ext_vector_type(4)))  __bf16 v4bf;
typedef __attribute__((ext_vector_type(8)))  float  v8f;

#define T_TOK 8192
#define DDIM  1024
#define EEXP  8
#define HDIM  1024
#define NSLOT (T_TOK * 2)
#define KSTAGE 64                 // K elements staged in LDS per buffer
#define NSTAGE (DDIM / KSTAGE)    // 16 (DDIM == HDIM so same for both GEMMs)

// ---------------------------------------------------------------- helpers
__device__ inline v8f wmma_bf16(v16bf a, v16bf b, v8f c) {
  return __builtin_amdgcn_wmma_f32_16x16x32_bf16(
      /*neg_a=*/false, a, /*neg_b=*/false, b,
      /*c_mod=*/(short)0, c, /*reuse_a=*/false, /*reuse_b=*/false);
}

// 16x32 bf16 fragment from LDS tile stored row-major [row][KSTAGE].
// Lane L<16: row L, K = {0..7, 16..23}; lane L>=16: row L-16, K = {8..15, 24..31}.
__device__ inline v16bf frag16(const bf16* tileRow0, int lane, int kOff) {
  const int r  = lane & 15;
  const int kb = kOff + ((lane >> 4) << 3);
  const v8bf lo = *(const v8bf*)(tileRow0 + r * KSTAGE + kb);
  const v8bf hi = *(const v8bf*)(tileRow0 + r * KSTAGE + kb + 16);
  v16bf f;
#pragma unroll
  for (int i = 0; i < 8; ++i) { f[i] = lo[i]; f[8 + i] = hi[i]; }
  return f;
}

__device__ inline v8bf zero8bf() {
  v8bf z;
#pragma unroll
  for (int i = 0; i < 8; ++i) z[i] = (bf16)0.0f;
  return z;
}

__device__ inline float silu_f(float g) { return g / (1.0f + __expf(-g)); }

// ---------------------------------------------------------------- utility kernels
__global__ __launch_bounds__(256) void zero_f32_kernel(float* __restrict__ p, size_t n4) {
  size_t i = ((size_t)blockIdx.x * 256 + threadIdx.x);
  if (i < n4) ((float4*)p)[i] = make_float4(0.f, 0.f, 0.f, 0.f);
}

__global__ void zero_i32_kernel(int* __restrict__ p, int n) {
  int i = blockIdx.x * 64 + threadIdx.x;
  if (i < n) p[i] = 0;
}

__global__ __launch_bounds__(256) void cvt_bf16_kernel(const float* __restrict__ src,
                                                       bf16* __restrict__ dst, size_t n4) {
  size_t i = ((size_t)blockIdx.x * 256 + threadIdx.x);
  if (i >= n4) return;
  float4 v = ((const float4*)src)[i];
  v4bf o;
  o[0] = (bf16)v.x; o[1] = (bf16)v.y; o[2] = (bf16)v.z; o[3] = (bf16)v.w;
  ((v4bf*)dst)[i] = o;
}

// ---------------------------------------------------------------- router
__global__ __launch_bounds__(256) void router_kernel(const float* __restrict__ x,
                                                     const float* __restrict__ wr,
                                                     int* __restrict__ counts,
                                                     int* __restrict__ tok_e,
                                                     float* __restrict__ tok_g) {
  const int t = blockIdx.x;
  const int lane = threadIdx.x & 31;
  const int e = threadIdx.x >> 5;
  const float4* xr = (const float4*)(x + (size_t)t * DDIM);
  const float4* wrow = (const float4*)(wr + (size_t)e * DDIM);
  float s = 0.f;
#pragma unroll 4
  for (int i = lane; i < DDIM / 4; i += 32) {
    float4 a = xr[i], b = wrow[i];
    s += a.x * b.x + a.y * b.y + a.z * b.z + a.w * b.w;
  }
#pragma unroll
  for (int m = 16; m; m >>= 1) s += __shfl_xor(s, m, 32);
  __shared__ float logits[EEXP];
  if (lane == 0) logits[e] = s;
  __syncthreads();
  if (threadIdx.x == 0) {
    int b0 = 0; float v0 = logits[0];
#pragma unroll
    for (int i = 1; i < EEXP; ++i) if (logits[i] > v0) { v0 = logits[i]; b0 = i; }
    int b1 = (b0 == 0) ? 1 : 0; float v1 = logits[b1];
#pragma unroll
    for (int i = 0; i < EEXP; ++i)
      if (i != b0 && logits[i] > v1) { v1 = logits[i]; b1 = i; }
    const float p0 = 1.0f / (1.0f + __expf(v1 - v0));
    const float p1 = 1.0f - p0;
    tok_e[t * 2 + 0] = b0; tok_e[t * 2 + 1] = b1;
    tok_g[t * 2 + 0] = p0; tok_g[t * 2 + 1] = p1;
    atomicAdd(&counts[b0], 1);
    atomicAdd(&counts[b1], 1);
  }
}

__global__ void offsets_kernel(const int* __restrict__ counts, int* __restrict__ offs) {
  if (threadIdx.x == 0) {
    int a = 0;
#pragma unroll
    for (int i = 0; i < EEXP; ++i) { offs[i] = a; a += counts[i]; }
  }
}

__global__ __launch_bounds__(256) void scatter_kernel(const int* __restrict__ tok_e,
                                                      const float* __restrict__ tok_g,
                                                      const int* __restrict__ offs,
                                                      int* __restrict__ cursors,
                                                      int* __restrict__ slot_tok,
                                                      float* __restrict__ slot_gate) {
  const int t = blockIdx.x * 256 + threadIdx.x;
  if (t >= T_TOK) return;
#pragma unroll
  for (int k = 0; k < 2; ++k) {
    const int e = tok_e[t * 2 + k];
    const int pos = atomicAdd(&cursors[e], 1);
    const int idx = offs[e] + pos;
    slot_tok[idx] = t;
    slot_gate[idx] = tok_g[t * 2 + k];
  }
}

// ---------------------------------------------------------------- expert GEMM 1 (x @ w_in^T, SwiGLU)
// grid: (token tiles, H/64, E); block 256 = 8 waves; wave -> 16x32 of 64x64 tile.
// Double-buffered LDS, K staged 64 at a time.
__global__ __launch_bounds__(256) void gemm1_kernel(const bf16* __restrict__ xb,
                                                    const bf16* __restrict__ wib,
                                                    const int* __restrict__ counts,
                                                    const int* __restrict__ offs,
                                                    const int* __restrict__ slot_tok,
                                                    bf16* __restrict__ hbuf) {
  const int e = blockIdx.z;
  const int cnt = counts[e];
  const int mBase = blockIdx.x * 64;
  if (mBase >= cnt) return;
  const int nBase = blockIdx.y * 64;
  const int off = offs[e];
  const bf16* wg = wib + (size_t)e * (2 * HDIM) * DDIM;

  __shared__ bf16 At[2][64 * KSTAGE];
  __shared__ bf16 Bg[2][64 * KSTAGE];
  __shared__ bf16 Bu[2][64 * KSTAGE];

  const int tid = threadIdx.x;
  const int lane = tid & 31;
  const int wv = tid >> 5;
  const int wm = (wv & 3) * 16;   // wave row offset in 64x64 tile
  const int wn = (wv >> 2) * 32;  // wave col offset

  v8f ag0 = {}, ag1 = {}, au0 = {}, au1 = {};

  // cooperative staging: thread covers row lrow, cols [cb, cb+16)
  const int lrow = tid >> 2;          // 0..63
  const int cb = (tid & 3) * 16;      // 0,16,32,48
  const bool arow_ok = (mBase + lrow < cnt);
  const bf16* asrc = nullptr;
  if (arow_ok) {
    const int tok = slot_tok[off + mBase + lrow];
    asrc = xb + (size_t)tok * DDIM + cb;
  }
  const bf16* bgsrc = wg + (size_t)(nBase + lrow) * DDIM + cb;
  const bf16* busrc = wg + (size_t)(HDIM + nBase + lrow) * DDIM + cb;
  bf16* const atDst = &At[0][lrow * KSTAGE + cb];
  bf16* const bgDst = &Bg[0][lrow * KSTAGE + cb];
  bf16* const buDst = &Bu[0][lrow * KSTAGE + cb];
  const int bufStride = 64 * KSTAGE;

  // ---- preload stage 0
  v8bf a0 = zero8bf(), a1 = zero8bf();
  if (arow_ok) { a0 = *(const v8bf*)(asrc); a1 = *(const v8bf*)(asrc + 8); }
  v8bf g0 = *(const v8bf*)(bgsrc),     g1 = *(const v8bf*)(bgsrc + 8);
  v8bf u0 = *(const v8bf*)(busrc),     u1 = *(const v8bf*)(busrc + 8);
  *(v8bf*)(atDst) = a0;  *(v8bf*)(atDst + 8) = a1;
  *(v8bf*)(bgDst) = g0;  *(v8bf*)(bgDst + 8) = g1;
  *(v8bf*)(buDst) = u0;  *(v8bf*)(buDst + 8) = u1;
  __syncthreads();

  for (int s = 0; s < NSTAGE; ++s) {
    const int cur = s & 1;
    const bool haveNext = (s + 1) < NSTAGE;
    // ---- issue next stage's global loads first (overlap with WMMA below)
    if (haveNext) {
      const int k0 = (s + 1) * KSTAGE;
      a0 = zero8bf(); a1 = zero8bf();
      if (arow_ok) { a0 = *(const v8bf*)(asrc + k0); a1 = *(const v8bf*)(asrc + k0 + 8); }
      g0 = *(const v8bf*)(bgsrc + k0); g1 = *(const v8bf*)(bgsrc + k0 + 8);
      u0 = *(const v8bf*)(busrc + k0); u1 = *(const v8bf*)(busrc + k0 + 8);
    }
    // ---- compute on current buffer
    const bf16* at = &At[cur][0];
    const bf16* bg = &Bg[cur][0];
    const bf16* bu = &Bu[cur][0];
#pragma unroll
    for (int ks = 0; ks < KSTAGE; ks += 32) {
      const v16bf af  = frag16(at + wm * KSTAGE, lane, ks);
      const v16bf bg0 = frag16(bg + wn * KSTAGE, lane, ks);
      const v16bf bg1 = frag16(bg + (wn + 16) * KSTAGE, lane, ks);
      const v16bf bu0 = frag16(bu + wn * KSTAGE, lane, ks);
      const v16bf bu1 = frag16(bu + (wn + 16) * KSTAGE, lane, ks);
      ag0 = wmma_bf16(af, bg0, ag0);
      ag1 = wmma_bf16(af, bg1, ag1);
      au0 = wmma_bf16(af, bu0, au0);
      au1 = wmma_bf16(af, bu1, au1);
    }
    // ---- stash next stage into the other buffer
    if (haveNext) {
      const int nxt = bufStride * ((s + 1) & 1);
      *(v8bf*)(atDst + nxt) = a0;  *(v8bf*)(atDst + nxt + 8) = a1;
      *(v8bf*)(bgDst + nxt) = g0;  *(v8bf*)(bgDst + nxt + 8) = g1;
      *(v8bf*)(buDst + nxt) = u0;  *(v8bf*)(buDst + nxt + 8) = u1;
    }
    __syncthreads();
  }

  // epilogue: h = silu(g) * u  -> bf16 hbuf[slot][H]
  const int cr = lane & 15;
  const int rg = (lane >> 4) * 8;
#pragma unroll
  for (int r = 0; r < 8; ++r) {
    const int m = wm + r + rg;
    if (mBase + m < cnt) {
      const int slot = off + mBase + m;
      bf16* dst = hbuf + (size_t)slot * HDIM + nBase + wn + cr;
      dst[0]  = (bf16)(silu_f(ag0[r]) * au0[r]);
      dst[16] = (bf16)(silu_f(ag1[r]) * au1[r]);
    }
  }
}

// ---------------------------------------------------------------- expert GEMM 2 (h @ w_out^T, gated combine)
__global__ __launch_bounds__(256) void gemm2_kernel(const bf16* __restrict__ hbuf,
                                                    const bf16* __restrict__ wob,
                                                    const int* __restrict__ counts,
                                                    const int* __restrict__ offs,
                                                    const int* __restrict__ slot_tok,
                                                    const float* __restrict__ slot_gate,
                                                    float* __restrict__ out) {
  const int e = blockIdx.z;
  const int cnt = counts[e];
  const int mBase = blockIdx.x * 64;
  if (mBase >= cnt) return;
  const int nBase = blockIdx.y * 64;  // over D
  const int off = offs[e];
  const bf16* w = wob + (size_t)e * DDIM * HDIM;

  __shared__ bf16 At[2][64 * KSTAGE];
  __shared__ bf16 Bt[2][64 * KSTAGE];

  const int tid = threadIdx.x;
  const int lane = tid & 31;
  const int wv = tid >> 5;
  const int wm = (wv & 3) * 16;
  const int wn = (wv >> 2) * 32;

  v8f ac0 = {}, ac1 = {};

  const int lrow = tid >> 2;
  const int cb = (tid & 3) * 16;
  const bool arow_ok = (mBase + lrow < cnt);
  const bf16* asrc = arow_ok ? (hbuf + (size_t)(off + mBase + lrow) * HDIM + cb) : nullptr;
  const bf16* bsrc = w + (size_t)(nBase + lrow) * HDIM + cb;
  bf16* const atDst = &At[0][lrow * KSTAGE + cb];
  bf16* const btDst = &Bt[0][lrow * KSTAGE + cb];
  const int bufStride = 64 * KSTAGE;

  v8bf a0 = zero8bf(), a1 = zero8bf();
  if (arow_ok) { a0 = *(const v8bf*)(asrc); a1 = *(const v8bf*)(asrc + 8); }
  v8bf b0v = *(const v8bf*)(bsrc), b1v = *(const v8bf*)(bsrc + 8);
  *(v8bf*)(atDst) = a0;  *(v8bf*)(atDst + 8) = a1;
  *(v8bf*)(btDst) = b0v; *(v8bf*)(btDst + 8) = b1v;
  __syncthreads();

  for (int s = 0; s < NSTAGE; ++s) {
    const int cur = s & 1;
    const bool haveNext = (s + 1) < NSTAGE;
    if (haveNext) {
      const int k0 = (s + 1) * KSTAGE;
      a0 = zero8bf(); a1 = zero8bf();
      if (arow_ok) { a0 = *(const v8bf*)(asrc + k0); a1 = *(const v8bf*)(asrc + k0 + 8); }
      b0v = *(const v8bf*)(bsrc + k0); b1v = *(const v8bf*)(bsrc + k0 + 8);
    }
    const bf16* at = &At[cur][0];
    const bf16* bt = &Bt[cur][0];
#pragma unroll
    for (int ks = 0; ks < KSTAGE; ks += 32) {
      const v16bf af = frag16(at + wm * KSTAGE, lane, ks);
      const v16bf f0 = frag16(bt + wn * KSTAGE, lane, ks);
      const v16bf f1 = frag16(bt + (wn + 16) * KSTAGE, lane, ks);
      ac0 = wmma_bf16(af, f0, ac0);
      ac1 = wmma_bf16(af, f1, ac1);
    }
    if (haveNext) {
      const int nxt = bufStride * ((s + 1) & 1);
      *(v8bf*)(atDst + nxt) = a0;  *(v8bf*)(atDst + nxt + 8) = a1;
      *(v8bf*)(btDst + nxt) = b0v; *(v8bf*)(btDst + nxt + 8) = b1v;
    }
    __syncthreads();
  }

  const int cr = lane & 15;
  const int rg = (lane >> 4) * 8;
#pragma unroll
  for (int r = 0; r < 8; ++r) {
    const int m = wm + r + rg;
    if (mBase + m < cnt) {
      const int slot = off + mBase + m;
      const int tok = slot_tok[slot];
      const float gate = slot_gate[slot];
      float* dst = out + (size_t)tok * DDIM + nBase + wn + cr;
      atomicAdd(dst +  0, gate * ac0[r]);
      atomicAdd(dst + 16, gate * ac1[r]);
    }
  }
}

// ---------------------------------------------------------------- launch
extern "C" void kernel_launch(void* const* d_in, const int* in_sizes, int n_in,
                              void* d_out, int out_size, void* d_ws, size_t ws_size,
                              hipStream_t stream) {
  (void)in_sizes; (void)n_in; (void)out_size; (void)ws_size;
  const float* x  = (const float*)d_in[0];   // [T, D]
  const float* wr = (const float*)d_in[1];   // [E, D]
  const float* wi = (const float*)d_in[2];   // [E, 2H, D]
  const float* wo = (const float*)d_in[3];   // [E, D, H]
  float* out = (float*)d_out;                // [T, D]

  uint8_t* p = (uint8_t*)d_ws;
  int*   counts    = (int*)p;                 p += 256;
  int*   cursors   = (int*)p;                 p += 256;
  int*   offs      = (int*)p;                 p += 256;
  int*   tok_e     = (int*)p;                 p += (size_t)T_TOK * 2 * sizeof(int);
  float* tok_g     = (float*)p;               p += (size_t)T_TOK * 2 * sizeof(float);
  int*   slot_tok  = (int*)p;                 p += (size_t)NSLOT * sizeof(int);
  float* slot_gate = (float*)p;               p += (size_t)NSLOT * sizeof(float);
  bf16*  xb        = (bf16*)p;                p += (size_t)T_TOK * DDIM * sizeof(bf16);
  bf16*  wib       = (bf16*)p;                p += (size_t)EEXP * 2 * HDIM * DDIM * sizeof(bf16);
  bf16*  wob       = (bf16*)p;                p += (size_t)EEXP * DDIM * HDIM * sizeof(bf16);
  bf16*  hbuf      = (bf16*)p;                p += (size_t)NSLOT * HDIM * sizeof(bf16);

  {
    const size_t n4 = (size_t)T_TOK * DDIM / 4;
    zero_f32_kernel<<<(unsigned)((n4 + 255) / 256), 256, 0, stream>>>(out, n4);
    zero_i32_kernel<<<1, 64, 0, stream>>>(counts, EEXP);
    zero_i32_kernel<<<1, 64, 0, stream>>>(cursors, EEXP);
  }
  {
    const size_t nx4 = (size_t)T_TOK * DDIM / 4;
    cvt_bf16_kernel<<<(unsigned)((nx4 + 255) / 256), 256, 0, stream>>>(x, xb, nx4);
    const size_t ni4 = (size_t)EEXP * 2 * HDIM * DDIM / 4;
    cvt_bf16_kernel<<<(unsigned)((ni4 + 255) / 256), 256, 0, stream>>>(wi, wib, ni4);
    const size_t no4 = (size_t)EEXP * DDIM * HDIM / 4;
    cvt_bf16_kernel<<<(unsigned)((no4 + 255) / 256), 256, 0, stream>>>(wo, wob, no4);
  }

  router_kernel<<<T_TOK, 256, 0, stream>>>(x, wr, counts, tok_e, tok_g);
  offsets_kernel<<<1, 32, 0, stream>>>(counts, offs);
  scatter_kernel<<<T_TOK / 256, 256, 0, stream>>>(tok_e, tok_g, offs, cursors,
                                                  slot_tok, slot_gate);

  {
    dim3 g1(T_TOK / 64, HDIM / 64, EEXP);
    gemm1_kernel<<<g1, 256, 0, stream>>>(xb, wib, counts, offs, slot_tok, hbuf);
    dim3 g2(T_TOK / 64, DDIM / 64, EEXP);
    gemm2_kernel<<<g2, 256, 0, stream>>>(hbuf, wob, counts, offs, slot_tok,
                                         slot_gate, out);
  }
}
